// LieGroupCore_18416819765804
// MI455X (gfx1250) — matrix operand order repaired and checked by reading, not compiled, not run on hardware
//
#include <hip/hip_runtime.h>
#include <hip/hip_bf16.h>

// ---------------------------------------------------------------------------
// Fused  lie_alg = z @ basis  (WMMA bf16x3, f32 accum)  +  expm(10x10 Taylor-8)
//
// MI455X reasoning: stage 1 is bound by reading z (256 MB @ 23.3 TB/s ~ 11us).
// bf16 WMMA compute for the GEMM (3x for the hi/lo error-compensated split) is
// ~5us, hidden under the loads. lie_alg never touches HBM (LDS-fused), saving
// 2x52MB. ||A|| <= ~0.3 so order-8 Taylor is exact to fp32 (err ~5e-11).
// This revision uses hardware f32<->bf16 converts ((__bf16) casts ->
// v_cvt_pk_bf16_f32) instead of manual RNE bit-twiddling to free VALU slots
// in the WMMA hot loop.
// ---------------------------------------------------------------------------

typedef __attribute__((ext_vector_type(16))) __bf16 v16bf;
typedef __attribute__((ext_vector_type(8)))  float  v8f;

#define ZDIM   512
#define NMAT   10
#define NCOL   100          // 10*10 flattened
#define NPAD   112          // padded to 7 WMMA N-tiles
#define NT     7            // N tiles of 16
#define KSTEPS 16           // 512 / 32
#define BFRAG_ELEMS (KSTEPS * NT * 32 * 16)   // 57344 bf16 per plane

// ---------------------------------------------------------------------------
// Pre-pack basis[512,100] -> zero-padded [512,112], split into bf16 hi/lo
// planes, laid out as per-lane B fragments (contiguous K order) so the main kernel
// loads each fragment as one aligned 32B vector per lane.
// B 32x16 fragment, contiguous K order (same as A): lane L holds column N=L%16; K = (L/16)*16 + e, e=0..15.
// ws layout: elem index = (((ks*7 + nt)*32 + lane)*16 + e); lo plane follows.
// ---------------------------------------------------------------------------
__global__ void pack_basis_kernel(const float* __restrict__ basis,
                                  __bf16* __restrict__ bws) {
    int tid = blockIdx.x * blockDim.x + threadIdx.x;   // one per (ks,nt,lane)
    if (tid >= KSTEPS * NT * 32) return;
    int lane = tid & 31;
    int nt   = (tid >> 5) % NT;
    int ks   = tid / (NT * 32);
    int n    = nt * 16 + (lane & 15);
    int kb   = ks * 32 + (lane >> 4) * 16;
    __bf16* hi = bws + (size_t)tid * 16;
    __bf16* lo = hi + BFRAG_ELEMS;
    #pragma unroll
    for (int e = 0; e < 16; ++e) {
        int k   = kb + e;
        float v = (n < NCOL) ? basis[(size_t)k * NCOL + n] : 0.0f;
        __bf16 h = (__bf16)v;            // hardware RNE convert
        hi[e] = h;
        lo[e] = (__bf16)(v - (float)h);  // residual, also RNE
    }
}

// ---------------------------------------------------------------------------
// Main fused kernel: 256 threads = 8 waves; wave w owns batch rows
// [blk*128 + w*16, +16). Stage 1: 7 f32 accumulators (16x16 each) over the
// padded 112 columns, K swept 32 at a time with v_wmma_f32_16x16x32_bf16
// (hi*hi + hi*lo + lo*hi). Stage 2: C tiles -> LDS, 2 threads/matrix Taylor.
// ---------------------------------------------------------------------------
#define LDS_STRIDE 113      // pad so column gathers hit distinct banks

__global__ __launch_bounds__(256)
void lie_expm_kernel(const float* __restrict__ z,
                     const __bf16* __restrict__ bws,
                     float* __restrict__ out) {
    __shared__ float tileC[128 * LDS_STRIDE];

    const int tid  = threadIdx.x;
    const int wave = tid >> 5;
    const int lane = tid & 31;
    const int g    = lane >> 4;       // K-group within fragment
    const int lm   = lane & 15;
    const long mbase = (long)blockIdx.x * 128 + wave * 16;
    const float* zrow = z + (mbase + lm) * ZDIM;

    const __bf16* bhi = bws;
    const __bf16* blo = bws + BFRAG_ELEMS;

    v8f acc[NT];
    const v8f zero8 = {0.f, 0.f, 0.f, 0.f, 0.f, 0.f, 0.f, 0.f};
    #pragma unroll
    for (int t = 0; t < NT; ++t) acc[t] = zero8;

    union V16 { v16bf v; __bf16 b[16]; };

    #pragma unroll 1
    for (int ks = 0; ks < KSTEPS; ++ks) {
        const int k0 = ks * 32;
        // A fragment: contiguous K order (matches packed B). Elements 0..15
        //             -> K = k0+g*16+0..15  (two 32B chunks)
        const float4* p0 = reinterpret_cast<const float4*>(zrow + k0 + g * 16);
        const float4* p1 = reinterpret_cast<const float4*>(zrow + k0 + g * 16 + 8);
        float4 f0 = p0[0], f1 = p0[1], f2 = p1[0], f3 = p1[1];
        __builtin_prefetch(zrow + k0 + 32, 0, 1);   // global_prefetch next chunk

        float buf[16] = {f0.x, f0.y, f0.z, f0.w, f1.x, f1.y, f1.z, f1.w,
                         f2.x, f2.y, f2.z, f2.w, f3.x, f3.y, f3.z, f3.w};
        V16 ah, al;
        float hv[16];
        #pragma unroll
        for (int e = 0; e < 16; ++e) {            // hi: packed hw converts
            __bf16 h = (__bf16)buf[e];
            ah.b[e] = h;
            hv[e]   = (float)h;
        }
        #pragma unroll
        for (int e = 0; e < 16; ++e)              // lo: residual converts
            al.b[e] = (__bf16)(buf[e] - hv[e]);

        #pragma unroll
        for (int t = 0; t < NT; ++t) {
            size_t off = ((size_t)(ks * NT + t) * 32 + lane) * 16;
            v16bf bh = *reinterpret_cast<const v16bf*>(bhi + off);
            v16bf bl = *reinterpret_cast<const v16bf*>(blo + off);
            // hi*hi + hi*lo + lo*hi  (bf16x3 error-compensated product)
            acc[t] = __builtin_amdgcn_wmma_f32_16x16x32_bf16(
                false, ah.v, false, bh, (short)0, acc[t], false, false);
            acc[t] = __builtin_amdgcn_wmma_f32_16x16x32_bf16(
                false, ah.v, false, bl, (short)0, acc[t], false, false);
            acc[t] = __builtin_amdgcn_wmma_f32_16x16x32_bf16(
                false, al.v, false, bh, (short)0, acc[t], false, false);
        }
    }

    // C layout: VGPR r, lanes 0-15: (M=r, N=lane); lanes 16-31: (M=8+r).
    #pragma unroll
    for (int t = 0; t < NT; ++t) {
        #pragma unroll
        for (int r = 0; r < 8; ++r) {
            int ml = g * 8 + r;
            tileC[(wave * 16 + ml) * LDS_STRIDE + t * 16 + lm] = acc[t][r];
        }
    }
    __syncthreads();

    // ---- Stage 2: expm via order-8 Taylor, 2 threads per matrix (5 rows) ----
    const int m  = tid >> 1;            // local matrix 0..127
    const int r0 = (tid & 1) * 5;       // this thread's first row
    const float* A = &tileC[m * LDS_STRIDE];   // entry (i,j) at i*10+j

    float T[5][NMAT], R[5][NMAT];
    #pragma unroll
    for (int i = 0; i < 5; ++i)
        #pragma unroll
        for (int j = 0; j < NMAT; ++j) {
            float a = A[(r0 + i) * NMAT + j];
            T[i][j] = a;
            R[i][j] = a + ((r0 + i) == j ? 1.0f : 0.0f);
        }

    #pragma unroll 1
    for (int o = 2; o <= 8; ++o) {
        float inv = 1.0f / (float)o;
        float tmp[5][NMAT];
        #pragma unroll
        for (int j = 0; j < NMAT; ++j) {
            float Ac[NMAT];
            #pragma unroll
            for (int k = 0; k < NMAT; ++k) Ac[k] = A[k * NMAT + j];
            #pragma unroll
            for (int i = 0; i < 5; ++i) {
                float s = 0.0f;
                #pragma unroll
                for (int k = 0; k < NMAT; ++k) s = fmaf(T[i][k], Ac[k], s);
                tmp[i][j] = s * inv;
            }
        }
        #pragma unroll
        for (int i = 0; i < 5; ++i)
            #pragma unroll
            for (int j = 0; j < NMAT; ++j) {
                T[i][j] = tmp[i][j];
                R[i][j] += tmp[i][j];
            }
    }

    float* dst = out + ((size_t)blockIdx.x * 128 + m) * NCOL + r0 * NMAT;
    #pragma unroll
    for (int i = 0; i < 5; ++i)
        #pragma unroll
        for (int j = 0; j < NMAT; ++j)
            dst[i * NMAT + j] = R[i][j];
}

extern "C" void kernel_launch(void* const* d_in, const int* in_sizes, int n_in,
                              void* d_out, int out_size, void* d_ws, size_t ws_size,
                              hipStream_t stream) {
    (void)n_in; (void)out_size; (void)ws_size;
    const float* z     = (const float*)d_in[0];
    const float* basis = (const float*)d_in[1];
    float* out         = (float*)d_out;
    __bf16* bws        = (__bf16*)d_ws;   // needs 2*57344*2 = 229376 B

    const int B = in_sizes[0] / ZDIM;              // 131072

    // Pre-swizzle basis into WMMA B-fragment order (hi/lo bf16 planes).
    const int packThreads = KSTEPS * NT * 32;      // 3584
    pack_basis_kernel<<<(packThreads + 255) / 256, 256, 0, stream>>>(basis, bws);

    // Fused GEMM (WMMA) + expm. 128 matrices per 256-thread block.
    lie_expm_kernel<<<B / 128, 256, 0, stream>>>(z, bws, out);
}
